// GATConv_1236950581780
// MI455X (gfx1250) — matrix operand order repaired and checked by reading, not compiled, not run on hardware
//
#include <hip/hip_runtime.h>
#include <math.h>

// ---------------- problem constants (match reference) ----------------
#define IN_FEATS   256
#define HEADS      8
#define OUT_FEATS  32
#define HF         256            // HEADS*OUT_FEATS
#define NEG_SLOPE  0.2f

// ---------------- WMMA vector types (gfx1250, wave32) ----------------
typedef __bf16 v16bf __attribute__((ext_vector_type(16)));
typedef float  v8f   __attribute__((ext_vector_type(8)));

// ---------------- helpers ----------------
// order-preserving float<->uint mapping for atomic max on floats
__device__ __forceinline__ unsigned f2ord(float f) {
  unsigned b = __float_as_uint(f);
  return (b & 0x80000000u) ? ~b : (b | 0x80000000u);
}
__device__ __forceinline__ float ord2f(unsigned u) {
  unsigned b = (u & 0x80000000u) ? (u & 0x7FFFFFFFu) : ~u;
  return __uint_as_float(b);
}
#define ORD_NEG_INF 0x007FFFFFu   // f2ord(-inf)

// split f32 into hi/lo bf16 (bf16x3 compensated product scheme)
__device__ __forceinline__ void splitbf(float f, __bf16& hi, __bf16& lo) {
  hi = (__bf16)f;
  lo = (__bf16)(f - (float)hi);
}

// =====================================================================
// Kernel 1: init output with bias (d_out is poisoned by harness)
// =====================================================================
__global__ void k_init_out(float* __restrict__ out, const float* __restrict__ bias,
                           int total) {
  int i = blockIdx.x * blockDim.x + threadIdx.x;
  if (i < total) out[i] = bias[i & (HF - 1)];
}

// Kernel 2: init segment-max (encoded -inf) and segment-sum (0)
__global__ void k_init_stats(unsigned* __restrict__ mm, float* __restrict__ ss, int nh) {
  int i = blockIdx.x * blockDim.x + threadIdx.x;
  if (i < nh) { mm[i] = ORD_NEG_INF; ss[i] = 0.0f; }
}

// =====================================================================
// Kernel 2b: one-time hi/lo bf16 split of W (removes all B-side
// conversion VALU from the GEMM inner loop; W' is L2-resident, 256 KB)
// =====================================================================
__global__ void k_split_w(const float* __restrict__ w,
                          __bf16* __restrict__ wh, __bf16* __restrict__ wl,
                          int total) {
  int i = blockIdx.x * blockDim.x + threadIdx.x;
  if (i >= total) return;
  float v = w[i];
  __bf16 h, l;
  splitbf(v, h, l);
  wh[i] = h; wl[i] = l;
}

// =====================================================================
// Kernel 3: h = feat @ W^T via v_wmma_f32_16x16x32_bf16 (bf16x3 split).
// One wave owns a 16-row block x 8 column tiles (128 cols): the A hi/lo
// fragments are loaded once per K-step and reused across all 8 tiles
// (feat read only 2x). B fragments come pre-split from wh/wl as direct
// v16bf loads -> inner loop is loads + WMMA only.
// =====================================================================
__global__ void k_gemm_wmma(const float* __restrict__ feat,
                            const __bf16* __restrict__ wh,   // [256 out][256 in] hi
                            const __bf16* __restrict__ wl,   // [256 out][256 in] lo
                            float* __restrict__ hbuf,        // [N][256]
                            int nNodes, int mTiles) {
  const int wid  = blockIdx.x * (blockDim.x >> 5) + (threadIdx.x >> 5);
  const int lane = threadIdx.x & 31;
  const int mTile = wid >> 1;
  const int nHalf = wid & 1;            // column tiles [0..7] or [8..15]
  if (mTile >= mTiles) return;          // wave-uniform exit
  const int mBase  = mTile * 16;
  const int nBase0 = nHalf * 128;

  const int  rrow = lane & 15;
  const bool hi   = lane >= 16;
  const bool full = (mBase + 16) <= nNodes;

  // A: lane = row M; K halves contiguous (0..15 | 16..31), matching B
  int arowi = mBase + rrow;
  if (arowi >= nNodes) arowi = nNodes - 1;        // clamp: OOB D-rows not stored
  const float* arow = feat + (size_t)arowi * IN_FEATS + (hi ? 16 : 0);
  // B: lane%16 = col N; lane/16 selects K half; W^T[k][n] = w[n][k]
  const size_t boff = (size_t)(nBase0 + rrow) * IN_FEATS + (hi ? 16 : 0);
  const __bf16* bh0 = wh + boff;
  const __bf16* bl0 = wl + boff;

  v8f acc[8] = {};

  for (int kb = 0; kb < IN_FEATS; kb += 32) {
    // ---- A fragment (shared across the 8 column tiles) ----
    const float4 a0 = *(const float4*)(arow + kb + 0);
    const float4 a1 = *(const float4*)(arow + kb + 4);
    const float4 a2 = *(const float4*)(arow + kb + 8);
    const float4 a3 = *(const float4*)(arow + kb + 12);
    float af[16] = {a0.x,a0.y,a0.z,a0.w, a1.x,a1.y,a1.z,a1.w,
                    a2.x,a2.y,a2.z,a2.w, a3.x,a3.y,a3.z,a3.w};
    v16bf Ah, Al;
#pragma unroll
    for (int i = 0; i < 16; ++i) { __bf16 h, l; splitbf(af[i], h, l); Ah[i] = h; Al[i] = l; }

#pragma unroll
    for (int t = 0; t < 8; ++t) {
      const size_t toff = (size_t)t * 16 * IN_FEATS + kb;
      const v16bf Bh = *(const v16bf*)(bh0 + toff);   // 32B aligned, 2x b128
      const v16bf Bl = *(const v16bf*)(bl0 + toff);
      // bf16x3: A*B ~= Ah*Bh + Ah*Bl + Al*Bh  (f32 accumulate)
      acc[t] = __builtin_amdgcn_wmma_f32_16x16x32_bf16(false, Ah, false, Bh,
                                                       (short)0, acc[t], false, false);
      acc[t] = __builtin_amdgcn_wmma_f32_16x16x32_bf16(false, Ah, false, Bl,
                                                       (short)0, acc[t], false, false);
      acc[t] = __builtin_amdgcn_wmma_f32_16x16x32_bf16(false, Al, false, Bh,
                                                       (short)0, acc[t], false, false);
    }
  }

  // D layout: VGPR v -> row v (+8 for lanes 16..31), col = lane%16
  if (full) {                     // fast path: straight-line stores, no guards
    float* op0 = hbuf + (size_t)(mBase + (hi ? 8 : 0)) * HF + nBase0 + rrow;
#pragma unroll
    for (int t = 0; t < 8; ++t) {
      float* op = op0 + t * 16;
#pragma unroll
      for (int v = 0; v < 8; ++v) op[(size_t)v * HF] = acc[t][v];
    }
  } else {                        // ragged tail tile
#pragma unroll
    for (int t = 0; t < 8; ++t) {
#pragma unroll
      for (int v = 0; v < 8; ++v) {
        int m = mBase + v + (hi ? 8 : 0);
        if (m < nNodes) hbuf[(size_t)m * HF + nBase0 + t * 16 + rrow] = acc[t][v];
      }
    }
  }
}

// =====================================================================
// Kernel 4: per-node attention logits el/er = <h[n,h,:], attn_{l,r}[h,:]>
// =====================================================================
__global__ void k_logits(const float* __restrict__ hbuf,
                         const float* __restrict__ al, const float* __restrict__ ar,
                         float* __restrict__ el, float* __restrict__ er, int nh) {
  int i = blockIdx.x * blockDim.x + threadIdx.x;
  if (i >= nh) return;
  int n = i >> 3, hh = i & 7;
  const float* hr  = hbuf + (size_t)n * HF + hh * OUT_FEATS;
  const float* alp = al + hh * OUT_FEATS;
  const float* arp = ar + hh * OUT_FEATS;
  float sl = 0.f, sr = 0.f;
#pragma unroll
  for (int f = 0; f < OUT_FEATS; ++f) { float x = hr[f]; sl += x * alp[f]; sr += x * arp[f]; }
  el[i] = sl; er[i] = sr;
}

// =====================================================================
// Kernel 5: segment max over destinations (ordered-uint atomicMax)
// =====================================================================
__global__ void k_edge_max(const float* __restrict__ el, const float* __restrict__ er,
                           const int* __restrict__ src, const int* __restrict__ dst,
                           unsigned* __restrict__ mm, int E) {
  int e = blockIdx.x * blockDim.x + threadIdx.x;
  if (e >= E) return;
  int s = src[e], d = dst[e];
  const float4 l0 = *(const float4*)(el + (size_t)s * HEADS);
  const float4 l1 = *(const float4*)(el + (size_t)s * HEADS + 4);
  const float4 r0 = *(const float4*)(er + (size_t)d * HEADS);
  const float4 r1 = *(const float4*)(er + (size_t)d * HEADS + 4);
  float x[8] = {l0.x + r0.x, l0.y + r0.y, l0.z + r0.z, l0.w + r0.w,
                l1.x + r1.x, l1.y + r1.y, l1.z + r1.z, l1.w + r1.w};
#pragma unroll
  for (int hh = 0; hh < HEADS; ++hh) {
    float v = x[hh] > 0.f ? x[hh] : NEG_SLOPE * x[hh];
    atomicMax(mm + (size_t)d * HEADS + hh, f2ord(v));
  }
}

// Kernel 6: decode max in place (uint->float, -inf -> 0 as in reference)
__global__ void k_finish_m(unsigned* __restrict__ mm, int nh) {
  int i = blockIdx.x * blockDim.x + threadIdx.x;
  if (i >= nh) return;
  float x = ord2f(mm[i]);
  if (!(x >= -3.0e38f && x <= 3.0e38f)) x = 0.f;   // -inf / NaN -> 0
  ((float*)mm)[i] = x;
}

// =====================================================================
// Kernel 7: segment sum of exp(e - m[dst])
// =====================================================================
__global__ void k_edge_expsum(const float* __restrict__ el, const float* __restrict__ er,
                              const int* __restrict__ src, const int* __restrict__ dst,
                              const float* __restrict__ mf, float* __restrict__ ss, int E) {
  int e = blockIdx.x * blockDim.x + threadIdx.x;
  if (e >= E) return;
  int s = src[e], d = dst[e];
  const float4 l0 = *(const float4*)(el + (size_t)s * HEADS);
  const float4 l1 = *(const float4*)(el + (size_t)s * HEADS + 4);
  const float4 r0 = *(const float4*)(er + (size_t)d * HEADS);
  const float4 r1 = *(const float4*)(er + (size_t)d * HEADS + 4);
  const float4 m0 = *(const float4*)(mf + (size_t)d * HEADS);
  const float4 m1 = *(const float4*)(mf + (size_t)d * HEADS + 4);
  float x[8] = {l0.x + r0.x, l0.y + r0.y, l0.z + r0.z, l0.w + r0.w,
                l1.x + r1.x, l1.y + r1.y, l1.z + r1.z, l1.w + r1.w};
  float m[8] = {m0.x, m0.y, m0.z, m0.w, m1.x, m1.y, m1.z, m1.w};
#pragma unroll
  for (int hh = 0; hh < HEADS; ++hh) {
    float v = x[hh] > 0.f ? x[hh] : NEG_SLOPE * x[hh];
    atomicAdd(ss + (size_t)d * HEADS + hh, __expf(v - m[hh]));
  }
}

// Kernel 8: reciprocal of sums (a = ex / where(s>0, s, 1))
__global__ void k_recip_s(float* __restrict__ ss, int nh) {
  int i = blockIdx.x * blockDim.x + threadIdx.x;
  if (i < nh) { float v = ss[i]; ss[i] = v > 0.f ? 1.f / v : 1.f; }
}

// =====================================================================
// Kernel 9: aggregation. One wave32 per edge; lanes 0..7 compute head
// weights, shfl-broadcast; per head a coalesced 128B load + 128B atomic.
// =====================================================================
__global__ void k_aggregate(const float* __restrict__ hbuf,
                            const float* __restrict__ el, const float* __restrict__ er,
                            const float* __restrict__ mf, const float* __restrict__ rs,
                            const int* __restrict__ src, const int* __restrict__ dst,
                            float* __restrict__ out, int E) {
  int w    = (blockIdx.x * blockDim.x + threadIdx.x) >> 5;
  int lane = threadIdx.x & 31;
  if (w >= E) return;                       // wave-uniform
  int s = src[w], d = dst[w];
  const float* hs = hbuf + (size_t)s * HF;
  __builtin_prefetch(hs, 0, 0);             // global_prefetch_b8 on the gathered row
  __builtin_prefetch(hs + 128, 0, 0);
  float a = 0.f;
  if (lane < HEADS) {
    float x = el[(size_t)s * HEADS + lane] + er[(size_t)d * HEADS + lane];
    x = x > 0.f ? x : NEG_SLOPE * x;
    a = __expf(x - mf[(size_t)d * HEADS + lane]) * rs[(size_t)d * HEADS + lane];
  }
  float* od = out + (size_t)d * HF;
#pragma unroll
  for (int hh = 0; hh < HEADS; ++hh) {
    float ah = __shfl(a, hh, 32);
    int f = (hh << 5) | lane;               // head*32 + feat
    atomicAdd(od + f, hs[f] * ah);
  }
}

// =====================================================================
// Launch
// =====================================================================
extern "C" void kernel_launch(void* const* d_in, const int* in_sizes, int n_in,
                              void* d_out, int out_size, void* d_ws, size_t ws_size,
                              hipStream_t stream) {
  const float* feat   = (const float*)d_in[0];
  const int*   src    = (const int*)  d_in[1];
  const int*   dst    = (const int*)  d_in[2];
  const float* fcw    = (const float*)d_in[3];
  const float* attn_l = (const float*)d_in[4];
  const float* attn_r = (const float*)d_in[5];
  const float* bias   = (const float*)d_in[6];
  float*       out    = (float*)d_out;

  const int N  = in_sizes[0] / IN_FEATS;
  const int E  = in_sizes[1];
  const int nh = N * HEADS;
  const int nw = in_sizes[3];               // 256*256 weight elements

  // ws layout: h [N*256 f32] | el [N*8] | er [N*8] | m [N*8 u32->f32] | s [N*8]
  //            | wh [64K bf16] | wl [64K bf16]
  char*     ws   = (char*)d_ws;
  float*    hbuf = (float*)ws;
  float*    el   = (float*)(ws + (size_t)N * HF * sizeof(float));
  float*    er   = el + (size_t)nh;
  unsigned* mm   = (unsigned*)(er + (size_t)nh);
  float*    mf   = (float*)mm;              // decoded in place by k_finish_m
  float*    ss   = (float*)(mm + (size_t)nh);
  __bf16*   wh   = (__bf16*)(ss + (size_t)nh);   // 64B-aligned offset
  __bf16*   wl   = wh + (size_t)nw;

  const int T = 256;
  const int totalOut = N * HF;

  k_init_out  <<<(totalOut + T - 1) / T, T, 0, stream>>>(out, bias, totalOut);
  k_init_stats<<<(nh + T - 1) / T,      T, 0, stream>>>(mm, ss, nh);
  k_split_w   <<<(nw + T - 1) / T,      T, 0, stream>>>(fcw, wh, wl, nw);

  const int mTiles = (N + 15) / 16;
  const int waves  = mTiles * 2;            // 2 column-halves per row block
  k_gemm_wmma <<<(waves + 7) / 8, T, 0, stream>>>(feat, wh, wl, hbuf, N, mTiles);

  k_logits    <<<(nh + T - 1) / T, T, 0, stream>>>(hbuf, attn_l, attn_r, el, er, nh);
  k_edge_max  <<<(E + T - 1) / T,  T, 0, stream>>>(el, er, src, dst, mm, E);
  k_finish_m  <<<(nh + T - 1) / T, T, 0, stream>>>(mm, nh);
  k_edge_expsum<<<(E + T - 1) / T, T, 0, stream>>>(el, er, src, dst, mf, ss, E);
  k_recip_s   <<<(nh + T - 1) / T, T, 0, stream>>>(ss, nh);
  k_aggregate <<<(E + 7) / 8,      T, 0, stream>>>(hbuf, el, er, mf, ss, src, dst, out, E);
}
